// AnchorModel_68951404970630
// MI455X (gfx1250) — matrix operand order repaired and checked by reading, not compiled, not run on hardware
//
#include <hip/hip_runtime.h>
#include <hip/hip_bf16.h>
#include <math.h>

// ---------------------------------------------------------------------------
// Problem constants (from the reference): inputs [N,L,H] f32, masks [N,L] f32,
// W [H,H] f32, b [H] f32  ->  out [N,N] f32 pairwise L2 of projected pooled emb
// ---------------------------------------------------------------------------
#define N_TOK 4096
#define SEQ_L 64
#define HDIM  512

typedef __attribute__((ext_vector_type(16))) __bf16 v16bf;
typedef __attribute__((ext_vector_type(8)))  float  v8f;

// ---------------------------------------------------------------------------
// WMMA wrapper: D = A(16x32 bf16) x B(32x16 bf16) + C(16x16 f32)
// ---------------------------------------------------------------------------
__device__ __forceinline__ v8f wmma_bf16(v16bf a, v16bf b, v8f c) {
    return __builtin_amdgcn_wmma_f32_16x16x32_bf16(
        /*neg_a=*/false, a, /*neg_b=*/false, b,
        /*c_mod=*/(short)0, c, /*reuse_a=*/false, /*reuse_b=*/false);
}

// ---------------------------------------------------------------------------
// Global-memory fragment loaders per CDNA5 ISA 7.12.2 VGPR layouts (wave32).
// A (16-bit, 16x32): lane m = lane&15, half = lane>>4; VGPR r holds K pair
// {Kbase, Kbase+1}, Kbase = 2*(r&3) + 8*half + 16*(r>>2).
// ---------------------------------------------------------------------------
__device__ __forceinline__ v16bf load_frag_A(const __bf16* __restrict__ mat,
                                             int row, int k0, int lane) {
    const unsigned int* p =
        (const unsigned int*)(mat + (size_t)row * HDIM + k0);
    const int half = lane >> 4;
    union { v16bf v; unsigned int u[8]; } F;
#pragma unroll
    for (int r = 0; r < 8; ++r) {
        const int widx = (r & 3) + 4 * half + 8 * (r >> 2); // K/2 word index
        F.u[r] = p[widx];
    }
    return F.v;
}

// B (16-bit, 32x16): lane n = lane&15, half = lane>>4; lane holds K values
// [8*half, 8*half+8) and [16+8*half, 24+8*half) of column n -> two b128 loads.
__device__ __forceinline__ v16bf load_frag_B(const __bf16* __restrict__ mat,
                                             int col, int k0, int lane) {
    const int half = lane >> 4;
    const uint4* q =
        (const uint4*)(mat + (size_t)col * HDIM + k0 + 8 * half);
    union { v16bf v; uint4 u[2]; } F;
    F.u[0] = q[0];
    F.u[1] = q[2];
    return F.v;
}

__device__ __forceinline__ void split_bf16(float x, __bf16& hi, __bf16& lo) {
    hi = (__bf16)x;
    lo = (__bf16)(x - (float)hi);
}

// ---------------------------------------------------------------------------
// K0: masked mean pooling; streams 512 MB of `inputs` with non-temporal loads
//     and emits pooled as split bf16 (hi/lo). One block per n, 256 threads.
// ---------------------------------------------------------------------------
__global__ __launch_bounds__(256) void pool_split_kernel(
    const float* __restrict__ inputs, const float* __restrict__ masks,
    __bf16* __restrict__ ph, __bf16* __restrict__ pl) {
    const int n = blockIdx.x;
    const int t = threadIdx.x;

    const float* mrow = masks + n * SEQ_L;
    float denom = 0.f;
#pragma unroll
    for (int l = 0; l < SEQ_L; ++l) denom += mrow[l];
    const float inv = 1.0f / denom;

    const float* in = inputs + (size_t)n * SEQ_L * HDIM;
    float acc0 = 0.f, acc1 = 0.f;
#pragma unroll 4
    for (int l = 0; l < SEQ_L; ++l) {
        const float m = mrow[l];
        acc0 += __builtin_nontemporal_load(&in[l * HDIM + t]) * m;
        acc1 += __builtin_nontemporal_load(&in[l * HDIM + t + 256]) * m;
    }
    acc0 *= inv;
    acc1 *= inv;

    __bf16 h0, l0, h1, l1;
    split_bf16(acc0, h0, l0);
    split_bf16(acc1, h1, l1);
    ph[(size_t)n * HDIM + t]       = h0;
    pl[(size_t)n * HDIM + t]       = l0;
    ph[(size_t)n * HDIM + t + 256] = h1;
    pl[(size_t)n * HDIM + t + 256] = l1;
}

// ---------------------------------------------------------------------------
// K1: W -> transposed split-bf16 (Wt[n][k] = W[k][n]).
// ---------------------------------------------------------------------------
__global__ __launch_bounds__(256) void wsplit_kernel(
    const float* __restrict__ W, __bf16* __restrict__ wth,
    __bf16* __restrict__ wtl) {
    const int idx = blockIdx.x * 256 + threadIdx.x; // over H*H
    const int k = idx / HDIM;
    const int n = idx % HDIM;
    __bf16 hi, lo;
    split_bf16(W[idx], hi, lo);
    wth[(size_t)n * HDIM + k] = hi;
    wtl[(size_t)n * HDIM + k] = lo;
}

// ---------------------------------------------------------------------------
// K2: projection emb = pooled @ W + b via split-bf16 WMMA, split-bf16 output.
//     Block tile 128x64, 8 waves, each a 32x32 subtile (2x2 WMMA tiles).
// ---------------------------------------------------------------------------
__global__ __launch_bounds__(256) void proj_kernel(
    const __bf16* __restrict__ ph, const __bf16* __restrict__ pl,
    const __bf16* __restrict__ wth, const __bf16* __restrict__ wtl,
    const float* __restrict__ bias, __bf16* __restrict__ eh,
    __bf16* __restrict__ el) {
    const int lane = threadIdx.x & 31;
    const int wave = threadIdx.x >> 5; // 0..7
    const int wr = wave >> 1;          // 0..3
    const int wc = wave & 1;           // 0..1
    const int rowbase = blockIdx.x * 128 + wr * 32;
    const int colbase = blockIdx.y * 64 + wc * 32;
    const int lm = lane & 15;

    v8f acc[2][2] = {};
    for (int k0 = 0; k0 < HDIM; k0 += 32) {
        v16bf ah[2], al[2], bh[2], bl[2];
#pragma unroll
        for (int t = 0; t < 2; ++t) {
            ah[t] = load_frag_A(ph, rowbase + t * 16 + lm, k0, lane);
            al[t] = load_frag_A(pl, rowbase + t * 16 + lm, k0, lane);
            bh[t] = load_frag_B(wth, colbase + t * 16 + lm, k0, lane);
            bl[t] = load_frag_B(wtl, colbase + t * 16 + lm, k0, lane);
        }
#pragma unroll
        for (int mt = 0; mt < 2; ++mt)
#pragma unroll
            for (int nt = 0; nt < 2; ++nt) {
                acc[mt][nt] = wmma_bf16(ah[mt], bh[nt], acc[mt][nt]);
                acc[mt][nt] = wmma_bf16(ah[mt], bl[nt], acc[mt][nt]);
                acc[mt][nt] = wmma_bf16(al[mt], bh[nt], acc[mt][nt]);
            }
    }

    const int half = lane >> 4;
#pragma unroll
    for (int mt = 0; mt < 2; ++mt)
#pragma unroll
        for (int nt = 0; nt < 2; ++nt) {
            const int j = colbase + nt * 16 + lm;
            const float bj = bias[j];
#pragma unroll
            for (int r = 0; r < 8; ++r) {
                const int i = rowbase + mt * 16 + r + 8 * half;
                const float v = acc[mt][nt][r] + bj;
                __bf16 hi, lo;
                split_bf16(v, hi, lo);
                eh[(size_t)i * HDIM + j] = hi;
                el[(size_t)i * HDIM + j] = lo;
            }
        }
}

// ---------------------------------------------------------------------------
// K3: squared row norms of emb (reconstructed hi+lo). One block per row.
// ---------------------------------------------------------------------------
__global__ __launch_bounds__(128) void norms_kernel(
    const __bf16* __restrict__ eh, const __bf16* __restrict__ el,
    float* __restrict__ sn) {
    const int row = blockIdx.x;
    const int t = threadIdx.x;
    __shared__ float red[128];
    float acc = 0.f;
#pragma unroll
    for (int k = t; k < HDIM; k += 128) {
        const float v = (float)eh[(size_t)row * HDIM + k] +
                        (float)el[(size_t)row * HDIM + k];
        acc += v * v;
    }
    red[t] = acc;
    __syncthreads();
#pragma unroll
    for (int s = 64; s > 0; s >>= 1) {
        if (t < s) red[t] += red[t + s];
        __syncthreads();
    }
    if (t == 0) sn[row] = red[0];
}

// ===========================================================================
// K4: Gram via split-bf16 WMMA + fused distance epilogue.
//     TDM (tensor_load_to_lds) double-buffered LDS staging when available.
// ===========================================================================
#if __has_builtin(__builtin_amdgcn_tensor_load_to_lds)
#define USE_TDM 1
#else
#define USE_TDM 0
#endif

#if USE_TDM
typedef unsigned int u32x4 __attribute__((ext_vector_type(4)));
typedef int          i32x4 __attribute__((ext_vector_type(4)));
typedef int          i32x8 __attribute__((ext_vector_type(8)));

// LDS tile geometry: 32 K-elems (64 B) per row + 16 B TDM pad -> 80 B stride.
// Padding makes half-wave row reads hit 16 distinct banks and keeps 16B align.
#define TROW_B   80
#define A_BYTES  (128 * TROW_B)                 // 10240
#define B_BYTES  (64 * TROW_B)                  // 5120
#define STAGE_B  (2 * A_BYTES + 2 * B_BYTES)    // 30720

// Build a 2D D# (ISA 8.3/8.4) and issue TENSOR_LOAD_TO_LDS.
//   tile = tile_rows x 32 bf16 slice of a row-major [*, HDIM] matrix.
//   g0: count=1 | lds_addr | global_addr[56:0] | type=2
//   g1: data_size=2B, pad_enable, pad_interval=16dw(64B), pad_amount=4dw(16B),
//       tensor_dim0=HDIM, tensor_dim1=N_TOK, tile_dim0=32, tile_dim1=rows,
//       tensor_dim0_stride=HDIM
__device__ __forceinline__ void tdm_load_tile(const void* gptr,
                                              unsigned lds_off,
                                              int tile_rows) {
    const unsigned long long ga = (unsigned long long)gptr;
    u32x4 g0;
    g0[0] = 1u;                                             // count=1, user
    g0[1] = lds_off;                                        // lds_addr
    g0[2] = (unsigned)(ga & 0xffffffffull);                 // global_addr lo
    g0[3] = (unsigned)((ga >> 32) & 0x1ffffffull)           // global_addr hi
            | (2u << 30);                                   // type=2 (image)
    i32x8 g1;
    g1[0] = (int)((1u << 16)        // data_size = 1 -> 2 bytes
                  | (1u << 20)      // pad_enable
                  | (3u << 22)      // pad_interval: code 3 -> 16 DWORDs (64B)
                  | (3u << 25));    // pad_amount:  code 3 -> 4 DWORDs (16B)
    g1[1] = (int)(((unsigned)HDIM & 0xffffu) << 16);        // tensor_dim0 lo16
    g1[2] = (int)(((unsigned)HDIM >> 16) |
                  (((unsigned)N_TOK & 0xffffu) << 16));     // dim0 hi | dim1 lo
    g1[3] = (int)(((unsigned)N_TOK >> 16) | (32u << 16));   // dim1 hi | tile_d0
    g1[4] = (int)((unsigned)tile_rows & 0xffffu);           // tile_dim1
    g1[5] = (int)(unsigned)HDIM;                            // dim0_stride lo32
    g1[6] = 0;                                              // stride hi | d1s
    g1[7] = 0;
    const i32x4 z4 = {0, 0, 0, 0};
    const i32x8 z8 = {0, 0, 0, 0, 0, 0, 0, 0};
    // 6-arg toolchain signature: (g0, g1, g2, g3, g_extra, cpol)
    __builtin_amdgcn_tensor_load_to_lds(g0, g1, z4, z4, z8, 0);
}

// LDS fragment loaders (same VGPR layouts, 80-byte row stride).
__device__ __forceinline__ v16bf lds_frag_A(const unsigned char* buf, int row,
                                            int lane) {
    const int half = lane >> 4;
    const unsigned int* p = (const unsigned int*)(buf + row * TROW_B);
    union { v16bf v; unsigned int u[8]; } F;
#pragma unroll
    for (int r = 0; r < 8; ++r)
        F.u[r] = p[(r & 3) + 4 * half + 8 * (r >> 2)];
    return F.v;
}

__device__ __forceinline__ v16bf lds_frag_B(const unsigned char* buf, int row,
                                            int lane) {
    const int half = lane >> 4;
    const uint4* q = (const uint4*)(buf + row * TROW_B + 16 * half);
    union { v16bf v; uint4 u[2]; } F;
    F.u[0] = q[0];
    F.u[1] = q[2];
    return F.v;
}
#endif // USE_TDM

__global__ __launch_bounds__(256) void gram_dist_kernel(
    const __bf16* __restrict__ eh, const __bf16* __restrict__ el,
    const float* __restrict__ sn, float* __restrict__ out) {
    const int lane = threadIdx.x & 31;
    const int wave = threadIdx.x >> 5;
    const int wr = wave >> 1;
    const int wc = wave & 1;
    const int rowblk = blockIdx.x * 128;
    const int colblk = blockIdx.y * 64;
    const int rowbase = rowblk + wr * 32;
    const int colbase = colblk + wc * 32;
    const int lm = lane & 15;

    v8f acc[2][2] = {};

#if USE_TDM
    __shared__ unsigned char smem[2 * STAGE_B];
    const unsigned lds0 = (unsigned)(unsigned long long)(void*)smem;

    auto issue_stage = [&](int s, int k0) {
        const unsigned base = lds0 + (unsigned)s * STAGE_B;
        tdm_load_tile(eh + (size_t)rowblk * HDIM + k0, base, 128);
        tdm_load_tile(el + (size_t)rowblk * HDIM + k0, base + A_BYTES, 128);
        tdm_load_tile(eh + (size_t)colblk * HDIM + k0, base + 2 * A_BYTES, 64);
        tdm_load_tile(el + (size_t)colblk * HDIM + k0,
                      base + 2 * A_BYTES + B_BYTES, 64);
    };

    if (wave == 0) issue_stage(0, 0);     // TDM ignores EXEC; once per wave

    for (int kk = 0; kk < HDIM / 32; ++kk) {
        const int cur = kk & 1;
        if (wave == 0) {
            if (kk + 1 < HDIM / 32) {
                issue_stage(cur ^ 1, (kk + 1) * 32); // overlap next-stage DMA
                __builtin_amdgcn_s_wait_tensorcnt(4); // in-order: cur done
            } else {
                __builtin_amdgcn_s_wait_tensorcnt(0);
            }
        }
        __syncthreads(); // stage `cur` visible to all waves

        const unsigned char* st = smem + cur * STAGE_B;
        v16bf ah[2], al[2], bh[2], bl[2];
#pragma unroll
        for (int t = 0; t < 2; ++t) {
            ah[t] = lds_frag_A(st, wr * 32 + t * 16 + lm, lane);
            al[t] = lds_frag_A(st + A_BYTES, wr * 32 + t * 16 + lm, lane);
            bh[t] = lds_frag_B(st + 2 * A_BYTES, wc * 32 + t * 16 + lm, lane);
            bl[t] = lds_frag_B(st + 2 * A_BYTES + B_BYTES,
                               wc * 32 + t * 16 + lm, lane);
        }
#pragma unroll
        for (int mt = 0; mt < 2; ++mt)
#pragma unroll
            for (int nt = 0; nt < 2; ++nt) {
                acc[mt][nt] = wmma_bf16(ah[mt], bh[nt], acc[mt][nt]);
                acc[mt][nt] = wmma_bf16(ah[mt], bl[nt], acc[mt][nt]);
                acc[mt][nt] = wmma_bf16(al[mt], bh[nt], acc[mt][nt]);
            }
        __syncthreads(); // all reads of `cur` done before it is re-filled
    }
#else
    for (int k0 = 0; k0 < HDIM; k0 += 32) {
        v16bf ah[2], al[2], bh[2], bl[2];
#pragma unroll
        for (int t = 0; t < 2; ++t) {
            ah[t] = load_frag_A(eh, rowbase + t * 16 + lm, k0, lane);
            al[t] = load_frag_A(el, rowbase + t * 16 + lm, k0, lane);
            bh[t] = load_frag_B(eh, colbase + t * 16 + lm, k0, lane);
            bl[t] = load_frag_B(el, colbase + t * 16 + lm, k0, lane);
        }
#pragma unroll
        for (int mt = 0; mt < 2; ++mt)
#pragma unroll
            for (int nt = 0; nt < 2; ++nt) {
                acc[mt][nt] = wmma_bf16(ah[mt], bh[nt], acc[mt][nt]);
                acc[mt][nt] = wmma_bf16(ah[mt], bl[nt], acc[mt][nt]);
                acc[mt][nt] = wmma_bf16(al[mt], bh[nt], acc[mt][nt]);
            }
    }
#endif

    // C layout: lane n = lane&15, half = lane>>4; VGPR r -> row m = r + 8*half
    const int half = lane >> 4;
#pragma unroll
    for (int mt = 0; mt < 2; ++mt)
#pragma unroll
        for (int nt = 0; nt < 2; ++nt) {
            const int j = colbase + nt * 16 + lm;
            const float snj = sn[j];
#pragma unroll
            for (int r = 0; r < 8; ++r) {
                const int i = rowbase + mt * 16 + r + 8 * half;
                float sq = sn[i] + snj - 2.0f * acc[mt][nt][r];
                out[(size_t)i * N_TOK + j] = (sq > 0.f) ? sqrtf(sq) : 0.f;
            }
        }
}

// ---------------------------------------------------------------------------
// Host-side orchestration
// ---------------------------------------------------------------------------
extern "C" void kernel_launch(void* const* d_in, const int* in_sizes, int n_in,
                              void* d_out, int out_size, void* d_ws,
                              size_t ws_size, hipStream_t stream) {
    (void)in_sizes; (void)n_in; (void)out_size; (void)ws_size;

    const float* inputs = (const float*)d_in[0]; // [N, L, H]
    const float* masks  = (const float*)d_in[1]; // [N, L]
    const float* W      = (const float*)d_in[2]; // [H, H]
    const float* bias   = (const float*)d_in[3]; // [H]
    float* out = (float*)d_out;                  // [N, N]

    // Workspace layout (256B aligned slices), ~18 MB total.
    char* ws = (char*)d_ws;
    size_t off = 0;
    auto take = [&](size_t bytes) -> char* {
        char* p = ws + off;
        off = (off + bytes + 255) & ~(size_t)255;
        return p;
    };
    __bf16* ph  = (__bf16*)take((size_t)N_TOK * HDIM * 2);
    __bf16* pl  = (__bf16*)take((size_t)N_TOK * HDIM * 2);
    __bf16* wth = (__bf16*)take((size_t)HDIM * HDIM * 2);
    __bf16* wtl = (__bf16*)take((size_t)HDIM * HDIM * 2);
    __bf16* eh  = (__bf16*)take((size_t)N_TOK * HDIM * 2);
    __bf16* el  = (__bf16*)take((size_t)N_TOK * HDIM * 2);
    float*  sn  = (float*)take((size_t)N_TOK * 4);

    // 1) masked mean pooling + bf16 split (streams 512 MB, NT loads)
    pool_split_kernel<<<N_TOK, 256, 0, stream>>>(inputs, masks, ph, pl);

    // 2) W transpose + bf16 split
    wsplit_kernel<<<(HDIM * HDIM) / 256, 256, 0, stream>>>(W, wth, wtl);

    // 3) projection via WMMA (split bf16), emits split emb
    proj_kernel<<<dim3(N_TOK / 128, HDIM / 64), 256, 0, stream>>>(
        ph, pl, wth, wtl, bias, eh, el);

    // 4) squared norms
    norms_kernel<<<N_TOK, 128, 0, stream>>>(eh, el, sn);

    // 5) Gram via WMMA (TDM-staged LDS tiles) + fused pairwise-L2 epilogue
    gram_dist_kernel<<<dim3(N_TOK / 128, N_TOK / 64), 256, 0, stream>>>(
        eh, el, sn, out);
}
